// MultiHeadAttentionBlock_42880953483484
// MI455X (gfx1250) — matrix operand order repaired and checked by reading, not compiled, not run on hardware
//
#include <hip/hip_runtime.h>
#include <hip/hip_bf16.h>

#define SEQ     2048
#define BATCH   4
#define NH      16
#define DKH     64
#define DMODEL  1024
#define MTOT    (BATCH * SEQ)   // 8192

typedef __attribute__((ext_vector_type(16))) __bf16 v16bf;
typedef __attribute__((ext_vector_type(8)))  float  v8f;

union Frag { v16bf v; uint4 q[2]; unsigned short u[16]; };
union Acc8 { v8f v; float f[8]; };

// round-to-nearest-even (used for result stores)
__device__ __forceinline__ unsigned short f2bf(float x) {
  union { float f; unsigned u; } c; c.f = x;
  unsigned u = c.u;
  u += 0x7FFFu + ((u >> 16) & 1u);
  return (unsigned short)(u >> 16);
}
// cheap round-half-up pack for LDS staging (fewer VALU ops)
__device__ __forceinline__ unsigned pk2(float a, float b) {
  union { float f; unsigned u; } x, y; x.f = a; y.f = b;
  return ((y.u + 0x8000u) & 0xFFFF0000u) | ((x.u + 0x8000u) >> 16);
}

// ---------------------------------------------------------------------------
// GEMM: Y[M,N] = A[M,K] * W[N,K]^T  (NT gemm, both K-contiguous)
// Double-buffered LDS software pipeline: global prefetch overlaps WMMA.
// TA = float (convert to bf16) or unsigned short (already bf16)
// LAYOUT 0: bf16 out [B,H,S,DK] | 1: bf16 out [B,H,DK,S] | 2: f32 out [M,N]
// ---------------------------------------------------------------------------
template <typename TA, int LAYOUT>
__global__ __launch_bounds__(256)
void gemm_bt_kernel(const TA* __restrict__ A, const float* __restrict__ W,
                    void* __restrict__ Y, float out_scale)
{
  constexpr int K = DMODEL;
  constexpr int PITCH = 40;  // 80B row pitch: odd multiple of 16B -> conflict-free b128
  __shared__ __align__(16) unsigned short ldsA[2][128 * PITCH];
  __shared__ __align__(16) unsigned short ldsB[2][128 * PITCH];

  const int tid  = threadIdx.x;
  const int wave = tid >> 5;
  const int lane = tid & 31;
  const int l16  = lane & 15;
  const int hi   = lane >> 4;

  const int bm = blockIdx.x * 128;
  const int bn = blockIdx.y * 128;
  const int wm = (wave & 1) * 64;   // 2 waves along M (64 rows each)
  const int wn = (wave >> 1) * 32;  // 4 waves along N (32 cols each)

  // per-thread staging coordinates (2 segments of 8 elements for A and B)
  const int row0 = (tid + 0)   >> 2;           //   0..63
  const int row1 = (tid + 256) >> 2;           //  64..127
  const int c8   = (tid & 3) << 3;             //  0,8,16,24

  Acc8 acc[4][2];
  #pragma unroll
  for (int i = 0; i < 4; ++i)
    #pragma unroll
    for (int j = 0; j < 2; ++j)
      #pragma unroll
      for (int v = 0; v < 8; ++v) acc[i][j].f[v] = 0.0f;

  // prefetch registers
  float4 pfa[2][2], pfb[2][2];
  uint4  pua[2];

  auto prefetch = [&](int kk) {
    const int rows[2] = {row0, row1};
    #pragma unroll
    for (int s = 0; s < 2; ++s) {
      if constexpr (sizeof(TA) == 4) {
        const float* sa = (const float*)A + (size_t)(bm + rows[s]) * K + kk + c8;
        pfa[s][0] = ((const float4*)sa)[0];
        pfa[s][1] = ((const float4*)sa)[1];
      } else {
        const TA* sa = A + (size_t)(bm + rows[s]) * K + kk + c8;
        pua[s] = *(const uint4*)sa;
      }
      const float* sb = W + (size_t)(bn + rows[s]) * K + kk + c8;
      pfb[s][0] = ((const float4*)sb)[0];
      pfb[s][1] = ((const float4*)sb)[1];
    }
  };
  auto stage = [&](int buf) {
    const int rows[2] = {row0, row1};
    #pragma unroll
    for (int s = 0; s < 2; ++s) {
      uint4 pa;
      if constexpr (sizeof(TA) == 4) {
        pa.x = pk2(pfa[s][0].x, pfa[s][0].y); pa.y = pk2(pfa[s][0].z, pfa[s][0].w);
        pa.z = pk2(pfa[s][1].x, pfa[s][1].y); pa.w = pk2(pfa[s][1].z, pfa[s][1].w);
      } else {
        pa = pua[s];
      }
      *(uint4*)&ldsA[buf][rows[s] * PITCH + c8] = pa;
      uint4 pb;
      pb.x = pk2(pfb[s][0].x, pfb[s][0].y); pb.y = pk2(pfb[s][0].z, pfb[s][0].w);
      pb.z = pk2(pfb[s][1].x, pfb[s][1].y); pb.w = pk2(pfb[s][1].z, pfb[s][1].w);
      *(uint4*)&ldsB[buf][rows[s] * PITCH + c8] = pb;
    }
  };

  prefetch(0);
  int cur = 0;
  for (int kk = 0; kk < K; kk += 32) {
    stage(cur);
    __syncthreads();
    if (kk + 32 < K) prefetch(kk + 32);  // loads in flight during WMMAs below

    // A 16x32 bf16 frag: lanes 0-15 K{0..7,16..23}, lanes 16-31 K{8..15,24..31}
    Frag afr[4], bfr[2];
    const int kb = hi * 8;
    #pragma unroll
    for (int i = 0; i < 4; ++i) {
      const unsigned short* p = &ldsA[cur][(wm + i * 16 + l16) * PITCH + kb];
      afr[i].q[0] = *(const uint4*)p;
      afr[i].q[1] = *(const uint4*)(p + 16);
    }
    // B 32x16 bf16 frag: lane = column N, lanes 0-15 K{0..7,16..23}, lanes 16-31 K{8..15,24..31}
    #pragma unroll
    for (int j = 0; j < 2; ++j) {
      const unsigned short* p = &ldsB[cur][(wn + j * 16 + l16) * PITCH + hi * 8];
      bfr[j].q[0] = *(const uint4*)p;
      bfr[j].q[1] = *(const uint4*)(p + 16);
    }
    #pragma unroll
    for (int i = 0; i < 4; ++i)
      #pragma unroll
      for (int j = 0; j < 2; ++j)
        acc[i][j].v = __builtin_amdgcn_wmma_f32_16x16x32_bf16(
            false, afr[i].v, false, bfr[j].v, (short)0, acc[i][j].v, false, false);
    cur ^= 1;
  }

  // epilogue: C/D layout VGPR v -> (M = v + hi*8, N = l16)
  #pragma unroll
  for (int i = 0; i < 4; ++i)
    #pragma unroll
    for (int j = 0; j < 2; ++j)
      #pragma unroll
      for (int v = 0; v < 8; ++v) {
        const int gm = bm + wm + i * 16 + v + hi * 8;
        const int gn = bn + wn + j * 16 + l16;
        const float val = acc[i][j].f[v] * out_scale;
        if constexpr (LAYOUT == 2) {
          ((float*)Y)[(size_t)gm * DMODEL + gn] = val;
        } else {
          const int b = gm >> 11, s = gm & (SEQ - 1);
          const int h = gn >> 6,  d = gn & (DKH - 1);
          size_t o;
          if constexpr (LAYOUT == 0)
            o = (((size_t)b * NH + h) * SEQ + s) * DKH + d;
          else
            o = (((size_t)b * NH + h) * DKH + d) * SEQ + s;
          ((unsigned short*)Y)[o] = f2bf(val);
        }
      }
}

// ---------------------------------------------------------------------------
// Flash attention, 64-key blocks: per (b,h), O = softmax(Q K^T) V.
// Q/K stored [BH, S, 64] bf16 (Q pre-scaled by 1/8), V stored [BH, 64, S] bf16.
// Workgroup = 8 waves = 128 query rows; each wave owns 16 query rows.
// Per iteration: 8 WMMAs for scores (4 16x16 tiles), one softmax update,
// 8 WMMAs for P*V (2 A-frags x 4 d-chunks) -> softmax overhead halved per key.
// ---------------------------------------------------------------------------
__global__ __launch_bounds__(256)
void attn_flash_kernel(const unsigned short* __restrict__ Qp,
                       const unsigned short* __restrict__ Kp,
                       const unsigned short* __restrict__ Vt,
                       unsigned short* __restrict__ X)
{
  constexpr int PP = 68;  // f32 pitch for 64 cols (272B = 17*16B, float4-aligned)
  __shared__ __align__(16) float ldsP[8 * 16 * PP];  // ~34.8KB

  const int tid  = threadIdx.x;
  const int wave = tid >> 5;
  const int lane = tid & 31;
  const int l16  = lane & 15;
  const int hi   = lane >> 4;

  const int bh    = blockIdx.y;                 // 0..63
  const int qbase = blockIdx.x * 128 + wave * 16;

  const unsigned short* Qb = Qp + (size_t)bh * SEQ * DKH;
  const unsigned short* Kb = Kp + (size_t)bh * SEQ * DKH;
  const unsigned short* Vb = Vt + (size_t)bh * DKH * SEQ;

  // Q fragments (row = qbase + l16), d-chunks 0..31 and 32..63, reused all blocks
  Frag qf[2];
  {
    const unsigned short* qr = Qb + (size_t)(qbase + l16) * DKH;
    const int kb = hi * 8;
    qf[0].q[0] = *(const uint4*)(qr + kb);
    qf[0].q[1] = *(const uint4*)(qr + kb + 16);
    qf[1].q[0] = *(const uint4*)(qr + 32 + kb);
    qf[1].q[1] = *(const uint4*)(qr + 32 + kb + 16);
  }

  Acc8 o[4];
  #pragma unroll
  for (int j = 0; j < 4; ++j)
    #pragma unroll
    for (int v = 0; v < 8; ++v) o[j].f[v] = 0.0f;

  float m[8], l[8];
  #pragma unroll
  for (int v = 0; v < 8; ++v) { m[v] = -1e30f; l[v] = 0.0f; }

  float* myP = &ldsP[wave * 16 * PP];

  for (int kp = 0; kp < SEQ; kp += 64) {
    // ---- scores: four 16x16 tiles over this 64-key block (8 WMMAs) ----
    Acc8 s[4];
    #pragma unroll
    for (int t = 0; t < 4; ++t) {
      Frag k0, k1;
      const unsigned short* kr = Kb + (size_t)(kp + t * 16 + l16) * DKH + hi * 8;
      k0.q[0] = *(const uint4*)(kr);      k0.q[1] = *(const uint4*)(kr + 16);
      k1.q[0] = *(const uint4*)(kr + 32); k1.q[1] = *(const uint4*)(kr + 48);
      Acc8 z;
      #pragma unroll
      for (int v = 0; v < 8; ++v) z.f[v] = 0.0f;
      s[t].v = __builtin_amdgcn_wmma_f32_16x16x32_bf16(false, qf[0].v, false, k0.v, (short)0, z.v,    false, false);
      s[t].v = __builtin_amdgcn_wmma_f32_16x16x32_bf16(false, qf[1].v, false, k1.v, (short)0, s[t].v, false, false);
    }

    // ---- online softmax (row = VGPR slot v + hi*8, spread over 16 lanes) ----
    float alpha[8];
    #pragma unroll
    for (int v = 0; v < 8; ++v) {
      float mx = fmaxf(fmaxf(s[0].f[v], s[1].f[v]), fmaxf(s[2].f[v], s[3].f[v]));
      mx = fmaxf(mx, __shfl_xor(mx, 1));
      mx = fmaxf(mx, __shfl_xor(mx, 2));
      mx = fmaxf(mx, __shfl_xor(mx, 4));
      mx = fmaxf(mx, __shfl_xor(mx, 8));
      const float mn = fmaxf(m[v], mx);
      alpha[v] = __expf(m[v] - mn);
      float p0 = __expf(s[0].f[v] - mn);
      float p1 = __expf(s[1].f[v] - mn);
      float p2 = __expf(s[2].f[v] - mn);
      float p3 = __expf(s[3].f[v] - mn);
      s[0].f[v] = p0; s[1].f[v] = p1; s[2].f[v] = p2; s[3].f[v] = p3;
      float rs = (p0 + p1) + (p2 + p3);
      rs += __shfl_xor(rs, 1);
      rs += __shfl_xor(rs, 2);
      rs += __shfl_xor(rs, 4);
      rs += __shfl_xor(rs, 8);
      l[v] = l[v] * alpha[v] + rs;
      m[v] = mn;
    }
    #pragma unroll
    for (int j = 0; j < 4; ++j)
      #pragma unroll
      for (int v = 0; v < 8; ++v) o[j].f[v] *= alpha[v];

    // ---- re-layout P (C/D form -> two A-fragments) via per-wave LDS ----
    #pragma unroll
    for (int t = 0; t < 4; ++t)
      #pragma unroll
      for (int v = 0; v < 8; ++v)
        myP[(v + hi * 8) * PP + t * 16 + l16] = s[t].f[v];
    asm volatile("s_wait_dscnt 0" ::: "memory");

    Frag pf[2];
    #pragma unroll
    for (int half = 0; half < 2; ++half) {
      const int kb = hi * 8;
      const float* pr = &myP[l16 * PP + half * 32 + kb];
      float t[16];
      *(float4*)&t[0]  = *(const float4*)(pr);
      *(float4*)&t[4]  = *(const float4*)(pr + 4);
      *(float4*)&t[8]  = *(const float4*)(pr + 16);
      *(float4*)&t[12] = *(const float4*)(pr + 20);
      #pragma unroll
      for (int e = 0; e < 16; ++e) pf[half].u[e] = f2bf(t[e]);
    }

    // ---- O += P[16x64] * V[64x16] per 16-wide d-chunk (8 WMMAs) ----
    #pragma unroll
    for (int half = 0; half < 2; ++half)
      #pragma unroll
      for (int j = 0; j < 4; ++j) {
        Frag vf;
        const unsigned short* vr =
            Vb + (size_t)(j * 16 + l16) * SEQ + kp + half * 32 + hi * 8;
        vf.q[0] = *(const uint4*)vr;
        vf.q[1] = *(const uint4*)(vr + 16);
        o[j].v = __builtin_amdgcn_wmma_f32_16x16x32_bf16(
            false, pf[half].v, false, vf.v, (short)0, o[j].v, false, false);
      }
  }

  // ---- normalize and store to X[B,S,1024] bf16 ----
  const int b = bh >> 4, h = bh & 15;
  #pragma unroll
  for (int v = 0; v < 8; ++v) l[v] = 1.0f / l[v];
  #pragma unroll
  for (int j = 0; j < 4; ++j)
    #pragma unroll
    for (int v = 0; v < 8; ++v) {
      const int s = qbase + v + hi * 8;
      const int c = h * DKH + j * 16 + l16;
      X[((size_t)(b * SEQ + s)) * DMODEL + c] = f2bf(o[j].f[v] * l[v]);
    }
}

// ---------------------------------------------------------------------------
extern "C" void kernel_launch(void* const* d_in, const int* in_sizes, int n_in,
                              void* d_out, int out_size, void* d_ws, size_t ws_size,
                              hipStream_t stream) {
  const float* q  = (const float*)d_in[0];
  const float* k  = (const float*)d_in[1];
  const float* v  = (const float*)d_in[2];
  const float* wq = (const float*)d_in[3];
  const float* wk = (const float*)d_in[4];
  const float* wv = (const float*)d_in[5];
  const float* wo = (const float*)d_in[6];

  const size_t NE = (size_t)BATCH * SEQ * DMODEL;  // 8,388,608 elems
  unsigned short* Qp = (unsigned short*)d_ws;      // bf16 [B,H,S,64]   16MB
  unsigned short* Kp = Qp + NE;                    // bf16 [B,H,S,64]   16MB
  unsigned short* Vt = Kp + NE;                    // bf16 [B,H,64,S]   16MB
  unsigned short* X  = Vt + NE;                    // bf16 [B,S,1024]   16MB

  dim3 gg(MTOT / 128, DMODEL / 128);  // (64, 8)
  // Q projection with 1/sqrt(d_k)=1/8 folded in
  gemm_bt_kernel<float, 0><<<gg, 256, 0, stream>>>(q, wq, Qp, 0.125f);
  gemm_bt_kernel<float, 0><<<gg, 256, 0, stream>>>(k, wk, Kp, 1.0f);
  gemm_bt_kernel<float, 1><<<gg, 256, 0, stream>>>(v, wv, Vt, 1.0f);

  attn_flash_kernel<<<dim3(SEQ / 128, BATCH * NH), 256, 0, stream>>>(Qp, Kp, Vt, X);

  gemm_bt_kernel<unsigned short, 2><<<gg, 256, 0, stream>>>(X, wo, (float*)d_out, 1.0f);
}
